// SelfAttention_67336497267166
// MI455X (gfx1250) — matrix operand order repaired and checked by reading, not compiled, not run on hardware
//
#include <hip/hip_runtime.h>

// Self-attention block for MI455X (gfx1250): GN -> QKV GEMM -> exact softmax
// attention with 256KB-LDS score tiles -> projection+residual. All GEMMs via
// v_wmma_f32_16x16x32_bf16 (wave32), bf16 data / f32 accumulate.

#define B_      8
#define C_      128
#define HW_     4096
#define SEQ_    4096
#define GROUPS_ 32
#define CPG_    4

typedef __attribute__((ext_vector_type(16))) __bf16 v16bf;
typedef __attribute__((ext_vector_type(8)))  __bf16 v8bf;
typedef __attribute__((ext_vector_type(8)))  float  v8f;

static __device__ __forceinline__ v16bf cat8(v8bf lo, v8bf hi) {
  return __builtin_shufflevector(lo, hi, 0, 1, 2, 3, 4, 5, 6, 7,
                                         8, 9, 10, 11, 12, 13, 14, 15);
}

// 16x32 bf16 A fragment. Memory: row-major [m][k], row_stride in elements.
// Lane layout (ISA 7.12.2): row = lane&15; lanes 0-15 hold K {0..7,16..23},
// lanes 16-31 hold K {8..15,24..31}.
static __device__ __forceinline__ v16bf load_a_frag(const __bf16* base, int row_stride,
                                                    int m0, int k0) {
  int lane = threadIdx.x & 31;
  int m  = m0 + (lane & 15);
  int kb = (lane & 16) ? 8 : 0;
  const __bf16* p = base + (size_t)m * row_stride + k0 + kb;
  v8bf lo = *(const v8bf*)(p);
  v8bf hi = *(const v8bf*)(p + 16);
  return cat8(lo, hi);
}

// 32x16 bf16 B fragment where memory holds Mat[n][k] row-major (K contiguous).
// Lane layout: col = lane&15; lanes 0-15 hold K {0..7,16..23}, lanes 16-31 K {8..15,24..31}.
static __device__ __forceinline__ v16bf load_b_frag(const __bf16* base, int row_stride,
                                                    int n0, int k0) {
  int lane = threadIdx.x & 31;
  int n  = n0 + (lane & 15);
  int kb = (lane & 16) ? 8 : 0;
  const __bf16* p = base + (size_t)n * row_stride + k0 + kb;
  v8bf lo = *(const v8bf*)(p);
  v8bf hi = *(const v8bf*)(p + 16);
  return cat8(lo, hi);
}

static __device__ __forceinline__ v8f wmma_bf16(v16bf a, v16bf b, v8f c) {
  // (neg_a, A, neg_b, B, c_mod, C, reuse_a, reuse_b)
  return __builtin_amdgcn_wmma_f32_16x16x32_bf16(false, a, false, b, (short)0, c,
                                                 false, false);
}

// ---------------------------------------------------------------------------
// GroupNorm: one block per (batch, group); writes h as bf16 pixel-major [HW,C]
// ---------------------------------------------------------------------------
__global__ __launch_bounds__(256) void gn_kernel(const float* __restrict__ x,
                                                 const float* __restrict__ gamma,
                                                 const float* __restrict__ beta,
                                                 __bf16* __restrict__ h) {
  int b = blockIdx.x >> 5;
  int g = blockIdx.x & 31;
  const float* xg = x + ((size_t)b * C_ + g * CPG_) * HW_;

  float s = 0.f, s2 = 0.f;
  for (int i = threadIdx.x; i < CPG_ * HW_; i += 256) {
    float v = xg[i];
    s += v;
    s2 += v * v;
  }
#pragma unroll
  for (int off = 16; off; off >>= 1) {
    s  += __shfl_xor(s, off, 32);
    s2 += __shfl_xor(s2, off, 32);
  }
  __shared__ float ls[8], ls2[8];
  if ((threadIdx.x & 31) == 0) {
    ls[threadIdx.x >> 5]  = s;
    ls2[threadIdx.x >> 5] = s2;
  }
  __syncthreads();
  float ts = 0.f, ts2 = 0.f;
#pragma unroll
  for (int i = 0; i < 8; ++i) {
    ts += ls[i];
    ts2 += ls2[i];
  }
  const float inv_n = 1.0f / (float)(CPG_ * HW_);
  float mean = ts * inv_n;
  float var  = ts2 * inv_n - mean * mean;
  float rstd = rsqrtf(var + 1e-5f);

  __bf16* hb = h + (size_t)b * HW_ * C_;
  for (int i = threadIdx.x; i < CPG_ * HW_; i += 256) {
    int cl = i >> 12;          // i / HW_
    int p  = i & (HW_ - 1);
    int c  = g * CPG_ + cl;
    float v = (xg[i] - mean) * rstd * gamma[c] + beta[c];
    hb[(size_t)p * C_ + c] = (__bf16)v;
  }
}

// ---------------------------------------------------------------------------
// Convert the four 128x128 fp32 weight matrices to bf16
// ---------------------------------------------------------------------------
__global__ __launch_bounds__(256) void wconv_kernel(const float* __restrict__ wq,
                                                    const float* __restrict__ wk,
                                                    const float* __restrict__ wv,
                                                    const float* __restrict__ wp,
                                                    __bf16* __restrict__ oq,
                                                    __bf16* __restrict__ ok,
                                                    __bf16* __restrict__ ov,
                                                    __bf16* __restrict__ op) {
  int i = blockIdx.x * 256 + threadIdx.x;
  if (i < C_ * C_) {
    oq[i] = (__bf16)wq[i];
    ok[i] = (__bf16)wk[i];
    ov[i] = (__bf16)wv[i];
    op[i] = (__bf16)wp[i];
  }
}

// ---------------------------------------------------------------------------
// QKV: per block one 16-pixel M tile; 8 waves cover the 8 N tiles (128 ch).
// q,k stored pixel-major [HW,C]; v stored channel-major [C,HW] so the PV
// B fragments are K-contiguous.
// ---------------------------------------------------------------------------
__global__ __launch_bounds__(256) void qkv_kernel(const __bf16* __restrict__ h,
                                                  const __bf16* __restrict__ wq,
                                                  const __bf16* __restrict__ wk,
                                                  const __bf16* __restrict__ wv,
                                                  const float* __restrict__ bq,
                                                  const float* __restrict__ bk,
                                                  const float* __restrict__ bv,
                                                  __bf16* __restrict__ q,
                                                  __bf16* __restrict__ k,
                                                  __bf16* __restrict__ v) {
  int wave = threadIdx.x >> 5;
  int lane = threadIdx.x & 31;
  int b  = blockIdx.x >> 8;           // HW_/16 == 256 tiles per batch
  int m0 = (blockIdx.x & 255) * 16;
  const __bf16* hb = h + (size_t)b * HW_ * C_;

  v16bf a[4];
#pragma unroll
  for (int ks = 0; ks < 4; ++ks) a[ks] = load_a_frag(hb, C_, m0, ks * 32);

  int n0 = wave * 16;
  int n  = n0 + (lane & 15);
  int mb = (lane & 16) ? 8 : 0;

  {  // Q
    v8f acc = {};
#pragma unroll
    for (int ks = 0; ks < 4; ++ks)
      acc = wmma_bf16(a[ks], load_b_frag(wq, C_, n0, ks * 32), acc);
    float bias = bq[n];
    __bf16* dst = q + (size_t)b * HW_ * C_;
#pragma unroll
    for (int r = 0; r < 8; ++r)
      dst[(size_t)(m0 + mb + r) * C_ + n] = (__bf16)(acc[r] + bias);
  }
  {  // K (pixel-major, same as Q)
    v8f acc = {};
#pragma unroll
    for (int ks = 0; ks < 4; ++ks)
      acc = wmma_bf16(a[ks], load_b_frag(wk, C_, n0, ks * 32), acc);
    float bias = bk[n];
    __bf16* dst = k + (size_t)b * HW_ * C_;
#pragma unroll
    for (int r = 0; r < 8; ++r)
      dst[(size_t)(m0 + mb + r) * C_ + n] = (__bf16)(acc[r] + bias);
  }
  {  // V (channel-major)
    v8f acc = {};
#pragma unroll
    for (int ks = 0; ks < 4; ++ks)
      acc = wmma_bf16(a[ks], load_b_frag(wv, C_, n0, ks * 32), acc);
    float bias = bv[n];
    __bf16* dst = v + (size_t)b * C_ * HW_;
#pragma unroll
    for (int r = 0; r < 8; ++r)
      dst[(size_t)n * HW_ + (m0 + mb + r)] = (__bf16)(acc[r] + bias);
  }
}

// ---------------------------------------------------------------------------
// Attention: one block per (batch, 16-query tile). Full score row-block
// S[16][4096] fp32 lives in 256KB of dynamic LDS (WGP has 320KB). Exact
// softmax in two LDS passes, then P@V with P re-read from LDS as bf16 A frags.
// ---------------------------------------------------------------------------
__global__ __launch_bounds__(256) void attn_kernel(const __bf16* __restrict__ q,
                                                   const __bf16* __restrict__ k,
                                                   const __bf16* __restrict__ v,
                                                   __bf16* __restrict__ o) {
  extern __shared__ float sS[];  // [16][SEQ_] fp32 scores = 256 KB
  __shared__ float sTmp[16][16];
  __shared__ float sMax[16];
  __shared__ float sSum[16];

  int wave = threadIdx.x >> 5;
  int lane = threadIdx.x & 31;
  int b  = blockIdx.x >> 8;
  int q0 = (blockIdx.x & 255) * 16;
  const __bf16* qb = q + (size_t)b * HW_ * C_;
  const __bf16* kb = k + (size_t)b * HW_ * C_;
  const __bf16* vb = v + (size_t)b * C_ * HW_;
  const float scale = 0.08838834764831845f;  // 1/sqrt(128)

  v16bf qa[4];
#pragma unroll
  for (int ks = 0; ks < 4; ++ks) qa[ks] = load_a_frag(qb, C_, q0, ks * 32);

  // ---- Phase 1: S = scale * Q K^T, D fragments scattered into LDS ----
  {
    int n  = lane & 15;
    int mb = (lane & 16) ? 8 : 0;
    for (int kt = wave; kt < SEQ_ / 16; kt += 8) {
      __builtin_prefetch(kb + (size_t)(kt + 8) * 16 * C_, 0, 0);  // global_prefetch_b8
      v8f acc = {};
#pragma unroll
      for (int ks = 0; ks < 4; ++ks)
        acc = wmma_bf16(qa[ks], load_b_frag(kb, C_, kt * 16, ks * 32), acc);
#pragma unroll
      for (int r = 0; r < 8; ++r)
        sS[(size_t)(mb + r) * SEQ_ + kt * 16 + n] = acc[r] * scale;
    }
  }
  __syncthreads();

  // ---- Softmax over each of the 16 rows (exact, two LDS passes) ----
  {
    int r   = threadIdx.x & 15;
    int seg = threadIdx.x >> 4;                 // 16 segments of 256 columns
    float* row = sS + (size_t)r * SEQ_ + seg * (SEQ_ / 16);
    float lmax = -3.0e38f;
    for (int i = 0; i < SEQ_ / 16; ++i) lmax = fmaxf(lmax, row[i]);
    sTmp[r][seg] = lmax;
    __syncthreads();
    if (threadIdx.x < 16) {
      float m = sTmp[threadIdx.x][0];
#pragma unroll
      for (int i = 1; i < 16; ++i) m = fmaxf(m, sTmp[threadIdx.x][i]);
      sMax[threadIdx.x] = m;
    }
    __syncthreads();
    float rmax = sMax[r];
    float lsum = 0.f;
    for (int i = 0; i < SEQ_ / 16; ++i) {
      float e = __expf(row[i] - rmax);
      row[i] = e;
      lsum += e;
    }
    sTmp[r][seg] = lsum;
    __syncthreads();
    if (threadIdx.x < 16) {
      float ssum = 0.f;
#pragma unroll
      for (int i = 0; i < 16; ++i) ssum += sTmp[threadIdx.x][i];
      sSum[threadIdx.x] = ssum;
    }
    __syncthreads();
  }

  // ---- Phase 2: O[16,128] = P[16,4096] @ V; wave owns one 16-channel tile ----
  {
    int n0  = wave * 16;
    int m   = lane & 15;
    int kbh = (lane & 16) ? 8 : 0;
    v8f acc = {};
    for (int kv0 = 0; kv0 < SEQ_; kv0 += 32) {
      __builtin_prefetch(vb + (size_t)(n0 + m) * HW_ + kv0 + 512, 0, 0);
      const float* rowp = sS + (size_t)m * SEQ_ + kv0 + kbh;
      v16bf pa;
#pragma unroll
      for (int e = 0; e < 8; ++e) pa[e] = (__bf16)rowp[e];
#pragma unroll
      for (int e = 0; e < 8; ++e) pa[8 + e] = (__bf16)rowp[16 + e];
      acc = wmma_bf16(pa, load_b_frag(vb, HW_, n0, kv0), acc);
    }
    int n  = lane & 15;
    int mb = (lane & 16) ? 8 : 0;
    __bf16* ob = o + (size_t)b * HW_ * C_;
#pragma unroll
    for (int r = 0; r < 8; ++r) {
      int mm = mb + r;
      float val = acc[r] / sSum[mm];
      ob[(size_t)(q0 + mm) * C_ + n0 + n] = (__bf16)val;
    }
  }
}

// ---------------------------------------------------------------------------
// Projection + residual: out[b,c,p] = x[b,c,p] + (O @ Wp^T)[p,c] + bp[c]
// ---------------------------------------------------------------------------
__global__ __launch_bounds__(256) void proj_kernel(const __bf16* __restrict__ o,
                                                   const __bf16* __restrict__ wp,
                                                   const float* __restrict__ bp,
                                                   const float* __restrict__ x,
                                                   float* __restrict__ out) {
  int wave = threadIdx.x >> 5;
  int lane = threadIdx.x & 31;
  int b  = blockIdx.x >> 8;
  int m0 = (blockIdx.x & 255) * 16;
  const __bf16* ob = o + (size_t)b * HW_ * C_;

  v16bf a[4];
#pragma unroll
  for (int ks = 0; ks < 4; ++ks) a[ks] = load_a_frag(ob, C_, m0, ks * 32);

  int n0 = wave * 16;
  v8f acc = {};
#pragma unroll
  for (int ks = 0; ks < 4; ++ks)
    acc = wmma_bf16(a[ks], load_b_frag(wp, C_, n0, ks * 32), acc);

  int n  = n0 + (lane & 15);
  int mb = (lane & 16) ? 8 : 0;
  float bias = bp[n];
  const float* xb = x + ((size_t)b * C_ + n) * HW_;
  float*       yb = out + ((size_t)b * C_ + n) * HW_;
#pragma unroll
  for (int r = 0; r < 8; ++r) {
    int p = m0 + mb + r;
    yb[p] = xb[p] + acc[r] + bias;
  }
}

// ---------------------------------------------------------------------------
extern "C" void kernel_launch(void* const* d_in, const int* in_sizes, int n_in,
                              void* d_out, int out_size, void* d_ws, size_t ws_size,
                              hipStream_t stream) {
  (void)in_sizes; (void)n_in; (void)out_size; (void)ws_size;
  const float* x    = (const float*)d_in[0];
  const float* gn_w = (const float*)d_in[1];
  const float* gn_b = (const float*)d_in[2];
  const float* wq   = (const float*)d_in[3];
  const float* bq   = (const float*)d_in[4];
  const float* wk   = (const float*)d_in[5];
  const float* bk   = (const float*)d_in[6];
  const float* wv   = (const float*)d_in[7];
  const float* bv   = (const float*)d_in[8];
  const float* wp   = (const float*)d_in[9];
  const float* bp   = (const float*)d_in[10];
  float* out = (float*)d_out;

  char* w = (char*)d_ws;
  auto carve = [&](size_t bytes) -> char* {
    char* p = w;
    w += (bytes + 255) & ~(size_t)255;
    return p;
  };
  const size_t tens = (size_t)B_ * HW_ * C_ * sizeof(__bf16);  // 8 MiB each
  __bf16* hbuf = (__bf16*)carve(tens);   // GN output, pixel-major
  __bf16* qbuf = (__bf16*)carve(tens);   // pixel-major [HW,C]
  __bf16* kbuf = (__bf16*)carve(tens);   // pixel-major [HW,C]
  __bf16* vbuf = (__bf16*)carve(tens);   // channel-major [C,HW]
  __bf16* obuf = (__bf16*)carve(tens);   // attention output, pixel-major
  __bf16* wqb  = (__bf16*)carve((size_t)C_ * C_ * sizeof(__bf16));
  __bf16* wkb  = (__bf16*)carve((size_t)C_ * C_ * sizeof(__bf16));
  __bf16* wvb  = (__bf16*)carve((size_t)C_ * C_ * sizeof(__bf16));
  __bf16* wpb  = (__bf16*)carve((size_t)C_ * C_ * sizeof(__bf16));

  gn_kernel<<<B_ * GROUPS_, 256, 0, stream>>>(x, gn_w, gn_b, hbuf);
  wconv_kernel<<<(C_ * C_ + 255) / 256, 256, 0, stream>>>(wq, wk, wv, wp,
                                                          wqb, wkb, wvb, wpb);
  qkv_kernel<<<B_ * (HW_ / 16), 256, 0, stream>>>(hbuf, wqb, wkb, wvb,
                                                  bq, bk, bv, qbuf, kbuf, vbuf);
  attn_kernel<<<B_ * (HW_ / 16), 256, 16 * SEQ_ * sizeof(float), stream>>>(
      qbuf, kbuf, vbuf, obuf);
  proj_kernel<<<B_ * (HW_ / 16), 256, 0, stream>>>(obuf, wpb, bp, x, out);
}
